// PixelGAT_17970143166991
// MI455X (gfx1250) — matrix operand order repaired and checked by reading, not compiled, not run on hardware
//
#include <hip/hip_runtime.h>
#include <hip/hip_bf16.h>

#define NNODES 100352
#define NEDGES 802816
#define EPRIME (NEDGES + NNODES)   // edges + self loops
#define NHEADS 4
#define NCH    32
#define FDIM   128                  // H*C
#define NGRAPH 2
#define KCHUNKS 4                   // 128 / 32
#define NTILES  8                   // 128 / 16 output column tiles
#define BPACK_USHORTS (NTILES * KCHUNKS * 32 * 16)   // 16384 bf16 = 32 KB

typedef __attribute__((ext_vector_type(16))) __bf16 v16bf;
typedef __attribute__((ext_vector_type(8)))  float  v8f;
typedef unsigned int u32x4 __attribute__((ext_vector_type(4)));
typedef int          i32x4 __attribute__((ext_vector_type(4)));
typedef int          i32x8 __attribute__((ext_vector_type(8)));

// TDM availability (arity differs: clang-22/ROCm7.2 = 5 args, clang-23 = 6 args)
#if defined(__has_builtin)
#  if __has_builtin(__builtin_amdgcn_tensor_load_to_lds) && __has_builtin(__builtin_amdgcn_s_wait_tensorcnt)
#    define HAVE_TDM 1
#  endif
#endif

union BF16Vec { v16bf v; unsigned short s[16]; uint4 q[2]; };

__device__ __forceinline__ unsigned short f2bf(float f) {
    unsigned b = __float_as_uint(f);
    unsigned r = b + 0x7FFFu + ((b >> 16) & 1u);   // round-to-nearest-even
    return (unsigned short)(r >> 16);
}

// monotonic float <-> uint mapping for atomicMax on floats
__device__ __forceinline__ unsigned fordu(float f) {
    unsigned b = __float_as_uint(f);
    return (b & 0x80000000u) ? ~b : (b | 0x80000000u);
}
__device__ __forceinline__ float uordf(unsigned u) {
    unsigned b = (u & 0x80000000u) ? (u ^ 0x80000000u) : ~u;
    return __uint_as_float(b);
}

// ---------------------------------------------------------------------------
// Layers 1/2 matmul: H = act @ W. act: [N,128] bf16 row-major, W pre-packed.
// Each wave owns one M-tile and computes the full 16x128 output strip.
// 8 named accumulators, 32 WMMAs. Packed B (32 KB) staged into LDS once per
// block via the Tensor Data Mover (TENSOR_LOAD_TO_LDS + s_wait_tensorcnt),
// made compiler-visible via an asm memory clobber, read back as ds_load_b128.
// ---------------------------------------------------------------------------
__global__ void matmul_l12_wmma(const unsigned short* __restrict__ Xbf,
                                const unsigned short* __restrict__ Bpack,
                                float* __restrict__ H) {
    __shared__ __align__(16) unsigned short lds_B[BPACK_USHORTS];

#if defined(HAVE_TDM)
    if (threadIdx.x < 32) {
        // D# group 0: count=1, lds_addr=0 (sole shared array), global_addr, type=2
        unsigned long long ga = (unsigned long long)Bpack;
        u32x4 g0 = { 1u, 0u, (unsigned)(ga & 0xffffffffu),
                     (unsigned)((ga >> 32) & 0x01ffffffu) | 0x80000000u };
        // D# group 1: data_size=1B, tensor_dim0=32768, tensor_dim1=1,
        //             tile_dim0=32768, tile_dim1=1, dim0_stride=32768
        i32x8 g1 = { 0, (int)0x80000000u, 0x00010000, (int)0x80000000u,
                     1, 32768, 0, 0 };
        i32x4 g2 = { 0, 0, 0, 0 };
        i32x4 g3 = { 0, 0, 0, 0 };
#if __clang_major__ >= 23
        i32x8 g4 = { 0, 0, 0, 0, 0, 0, 0, 0 };
        __builtin_amdgcn_tensor_load_to_lds(g0, g1, g2, g3, g4, 0);
#else
        __builtin_amdgcn_tensor_load_to_lds(g0, g1, g2, g3, 0);
#endif
        __builtin_amdgcn_s_wait_tensorcnt(0);
    }
    __syncthreads();
#else
    // fallback: cooperative staging, 2048 x b128
    for (int i = threadIdx.x; i < BPACK_USHORTS / 8; i += 256)
        ((uint4*)lds_B)[i] = ((const uint4*)Bpack)[i];
    __syncthreads();
#endif
    // The TDM writes LDS through a descriptor the compiler cannot see.
    // Escape the LDS pointer + memory clobber so the subsequent ds loads
    // are not folded away as reads of never-written memory.
    asm volatile("" : : "r"(&lds_B[0]) : "memory");

    const int wave = threadIdx.x >> 5;
    const int lane = threadIdx.x & 31;
    const bool hiHalf = lane >= 16;
    const int l15 = lane & 15;
    const int mtile = blockIdx.x * 8 + wave;       // one M-tile per wave
    const int arow = mtile * 16 + l15;
    const unsigned short* xrow = Xbf + (long long)arow * FDIM;

    v8f acc0 = {}, acc1 = {}, acc2 = {}, acc3 = {};
    v8f acc4 = {}, acc5 = {}, acc6 = {}, acc7 = {};

#define NT_STEP(NT, ACC)                                                       \
    {                                                                          \
        BF16Vec b;                                                             \
        const unsigned short* bp =                                             \
            lds_B + ((((NT) * KCHUNKS) + kc) * 32 + lane) * 16;                \
        b.q[0] = *(const uint4*)(bp);                                          \
        b.q[1] = *(const uint4*)(bp + 8);                                      \
        ACC = __builtin_amdgcn_wmma_f32_16x16x32_bf16(                         \
            false, a.v, false, b.v, (short)0, ACC, false, false);              \
    }

#pragma unroll
    for (int kc = 0; kc < KCHUNKS; ++kc) {
        BF16Vec a;
        const int kbaseA = kc * 32 + (hiHalf ? 16 : 0);
        a.q[0] = *(const uint4*)(xrow + kbaseA);        // K = kbaseA .. +7
        a.q[1] = *(const uint4*)(xrow + kbaseA + 8);    // K = kbaseA+8 .. +15
        NT_STEP(0, acc0)
        NT_STEP(1, acc1)
        NT_STEP(2, acc2)
        NT_STEP(3, acc3)
        NT_STEP(4, acc4)
        NT_STEP(5, acc5)
        NT_STEP(6, acc6)
        NT_STEP(7, acc7)
    }
#undef NT_STEP

    const int rbase = mtile * 16 + (hiHalf ? 8 : 0);
    float* hbase = H + (long long)rbase * FDIM + l15;

#define NT_STORE(NT, ACC)                                                      \
    {                                                                          \
        float* hp = hbase + (NT) * 16;                                         \
        hp[0 * FDIM] = ACC[0]; hp[1 * FDIM] = ACC[1];                          \
        hp[2 * FDIM] = ACC[2]; hp[3 * FDIM] = ACC[3];                          \
        hp[4 * FDIM] = ACC[4]; hp[5 * FDIM] = ACC[5];                          \
        hp[6 * FDIM] = ACC[6]; hp[7 * FDIM] = ACC[7];                          \
    }
    NT_STORE(0, acc0)
    NT_STORE(1, acc1)
    NT_STORE(2, acc2)
    NT_STORE(3, acc3)
    NT_STORE(4, acc4)
    NT_STORE(5, acc5)
    NT_STORE(6, acc6)
    NT_STORE(7, acc7)
#undef NT_STORE
}

// ---------------------------------------------------------------------------
// Layer 0 matmul: H = X @ W0, X: [N,3] f32, W0: [3,128] f32. K padded to 32.
// ---------------------------------------------------------------------------
__global__ void matmul_l0_wmma(const float* __restrict__ X,
                               const float* __restrict__ W,
                               float* __restrict__ H) {
    const int wave = threadIdx.x >> 5;
    const int lane = threadIdx.x & 31;
    const int mtile = blockIdx.x;
    const bool hiHalf = lane >= 16;
    const int l15 = lane & 15;
    const int arow = mtile * 16 + l15;
    const int bcol = wave * 16 + l15;

    BF16Vec a, b;
#pragma unroll
    for (int i = 0; i < 16; ++i) { a.s[i] = 0; b.s[i] = 0; }
    if (!hiHalf) {
        const float* xp = X + (long long)arow * 3;
        a.s[0] = f2bf(xp[0]); a.s[1] = f2bf(xp[1]); a.s[2] = f2bf(xp[2]);
        b.s[0] = f2bf(W[0 * FDIM + bcol]);
        b.s[1] = f2bf(W[1 * FDIM + bcol]);
        b.s[2] = f2bf(W[2 * FDIM + bcol]);
    }
    v8f acc = {};
    acc = __builtin_amdgcn_wmma_f32_16x16x32_bf16(
        false, a.v, false, b.v, (short)0, acc, false, false);

    const int rbase = mtile * 16 + (hiHalf ? 8 : 0);
#pragma unroll
    for (int r = 0; r < 8; ++r)
        H[(long long)(rbase + r) * FDIM + bcol] = acc[r];
}

// ---------------------------------------------------------------------------
// Pack W [128,128] f32 row-major into the B-operand lane order matmul_l12 uses (bf16, contiguous k = 16h + i)
// ---------------------------------------------------------------------------
__global__ void pack_W_bf16(const float* __restrict__ W,
                            unsigned short* __restrict__ Bpack) {
    int t = blockIdx.x * blockDim.x + threadIdx.x;
    if (t >= NTILES * KCHUNKS * 32) return;
    int lane = t & 31;
    int kc   = (t >> 5) & 3;
    int nt   = t >> 7;
    int col   = nt * 16 + (lane & 15);
    int kbase = kc * 32 + ((lane & 16) ? 16 : 0);
    unsigned short* outp = Bpack + (unsigned)t * 16;
#pragma unroll
    for (int i = 0; i < 16; ++i)
        outp[i] = f2bf(W[(kbase + i) * FDIM + col]);
}

// a_src[n,h] = <h[n,h,:], att_src[h,:]>, a_dst likewise. one thread per (n,h)
__global__ void gat_attn_coef(const float* __restrict__ H,
                              const float* __restrict__ att_src,
                              const float* __restrict__ att_dst,
                              float* __restrict__ a_src,
                              float* __restrict__ a_dst) {
    int t = blockIdx.x * blockDim.x + threadIdx.x;
    if (t >= NNODES * NHEADS) return;
    int n = t >> 2, hh = t & 3;
    const float* hp = H + (long long)n * FDIM + hh * NCH;
    const float* as = att_src + hh * NCH;
    const float* ad = att_dst + hh * NCH;
    float s = 0.f, d = 0.f;
#pragma unroll
    for (int c4 = 0; c4 < NCH; c4 += 4) {
        float4 hv = *(const float4*)(hp + c4);
        float4 sv = *(const float4*)(as + c4);
        float4 dv = *(const float4*)(ad + c4);
        s += hv.x * sv.x + hv.y * sv.y + hv.z * sv.z + hv.w * sv.w;
        d += hv.x * dv.x + hv.y * dv.y + hv.z * dv.z + hv.w * dv.w;
    }
    a_src[t] = s;
    a_dst[t] = d;
}

__device__ __forceinline__ void edge_sd(int t, const int* __restrict__ ei, int& s, int& d) {
    if (t < NEDGES) { s = ei[t]; d = ei[NEDGES + t]; }
    else            { s = t - NEDGES; d = t - NEDGES; }
}

__device__ __forceinline__ float lrelu(float e) { return e > 0.f ? e : 0.2f * e; }

// pass 1: segment max of leakyrelu(a_src[s]+a_dst[d]) into m[dst] (ordered-uint)
__global__ void edge_max_k(const int* __restrict__ ei,
                           const float* __restrict__ a_src,
                           const float* __restrict__ a_dst,
                           unsigned* __restrict__ m) {
    int t = blockIdx.x * blockDim.x + threadIdx.x;
    if (t >= EPRIME) return;
    int s, d; edge_sd(t, ei, s, d);
    float4 as = *(const float4*)(a_src + (long long)s * 4);
    float4 ad = *(const float4*)(a_dst + (long long)d * 4);
    unsigned* mp = m + (long long)d * 4;
    atomicMax(mp + 0, fordu(lrelu(as.x + ad.x)));
    atomicMax(mp + 1, fordu(lrelu(as.y + ad.y)));
    atomicMax(mp + 2, fordu(lrelu(as.z + ad.z)));
    atomicMax(mp + 3, fordu(lrelu(as.w + ad.w)));
}

// pass 2: ex = exp(e - m[dst]); den[dst] += ex; store ex per edge
__global__ void edge_exp_k(const int* __restrict__ ei,
                           const float* __restrict__ a_src,
                           const float* __restrict__ a_dst,
                           const unsigned* __restrict__ m,
                           float* __restrict__ den,
                           float* __restrict__ ex) {
    int t = blockIdx.x * blockDim.x + threadIdx.x;
    if (t >= EPRIME) return;
    int s, d; edge_sd(t, ei, s, d);
    float4 as = *(const float4*)(a_src + (long long)s * 4);
    float4 ad = *(const float4*)(a_dst + (long long)d * 4);
    const unsigned* mp = m + (long long)d * 4;
    float4 v;
    v.x = __expf(lrelu(as.x + ad.x) - uordf(mp[0]));
    v.y = __expf(lrelu(as.y + ad.y) - uordf(mp[1]));
    v.z = __expf(lrelu(as.z + ad.z) - uordf(mp[2]));
    v.w = __expf(lrelu(as.w + ad.w) - uordf(mp[3]));
    *(float4*)(ex + (long long)t * 4) = v;
    float* dp = den + (long long)d * 4;
    atomicAdd(dp + 0, v.x);
    atomicAdd(dp + 1, v.y);
    atomicAdd(dp + 2, v.z);
    atomicAdd(dp + 3, v.w);
}

// pass 3: out[dst] += h[src] * alpha. one wave per edge, 4 channels per lane.
__global__ void edge_agg_k(const int* __restrict__ ei,
                           const float* __restrict__ H,
                           const float* __restrict__ ex,
                           const float* __restrict__ den,
                           float* __restrict__ outp) {
    unsigned long long t = (unsigned long long)blockIdx.x * blockDim.x + threadIdx.x;
    int edge = (int)(t >> 5);
    if (edge >= EPRIME) return;
    int lane = (int)(t & 31);
    int s, d; edge_sd(edge, ei, s, d);
    int hc = lane * 4;           // channels hc..hc+3, all in head hc/32
    int hh = lane >> 3;
    float alpha = ex[(long long)edge * 4 + hh] / (den[(long long)d * 4 + hh] + 1e-16f);
    const float4 hv = *(const float4*)(H + (long long)s * FDIM + hc);
    float* op = outp + (long long)d * FDIM + hc;
    atomicAdd(op + 0, hv.x * alpha);
    atomicAdd(op + 1, hv.y * alpha);
    atomicAdd(op + 2, hv.z * alpha);
    atomicAdd(op + 3, hv.w * alpha);
}

// concat layers: act = elu(out + bias) stored directly as bf16 for next WMMA
__global__ void act_elu_bias_bf16(const float* __restrict__ outp,
                                  const float* __restrict__ bias,
                                  unsigned short* __restrict__ act) {
    long long t = (long long)blockIdx.x * blockDim.x + threadIdx.x;
    if (t >= (long long)NNODES * FDIM) return;
    int hc = (int)(t & (FDIM - 1));
    float v = outp[t] + bias[hc];
    v = v > 0.f ? v : (__expf(v) - 1.f);
    act[t] = f2bf(v);
}

// final layer: head-mean + b2, then global mean-pool accumulation per graph
__global__ void final_pool(const float* __restrict__ outp,
                           const float* __restrict__ b2,
                           const int* __restrict__ batch,
                           float* __restrict__ psum,
                           float* __restrict__ pcnt) {
    int t = blockIdx.x * blockDim.x + threadIdx.x;
    if (t >= NNODES * NCH) return;
    int n = t >> 5, c = t & 31;
    const float* op = outp + (long long)n * FDIM;
    float v = 0.25f * (op[c] + op[NCH + c] + op[2 * NCH + c] + op[3 * NCH + c]) + b2[c];
    int g = batch[n];
    atomicAdd(&psum[g * NCH + c], v);
    if (c == 0) atomicAdd(&pcnt[g], 1.0f);
}

// tiny MLP head: [G,32] -> relu(@W1+b1) -> @W2+b2 -> [G,10]. single block.
__global__ void mlp_head(const float* __restrict__ psum,
                         const float* __restrict__ pcnt,
                         const float* __restrict__ W1, const float* __restrict__ b1,
                         const float* __restrict__ W2, const float* __restrict__ b2,
                         float* __restrict__ outp) {
    __shared__ float hid[NGRAPH * NCH];
    int t = threadIdx.x;              // 64 threads
    int g = t >> 5, c = t & 31;
    float cnt = pcnt[g]; cnt = cnt > 1.f ? cnt : 1.f;
    float acc = b1[c];
#pragma unroll 8
    for (int k = 0; k < NCH; ++k) acc += (psum[g * NCH + k] / cnt) * W1[k * NCH + c];
    hid[t] = acc > 0.f ? acc : 0.f;
    __syncthreads();
    if (t < NGRAPH * 10) {
        int gg = t / 10, o = t % 10;
        float a = b2[o];
#pragma unroll 8
        for (int k = 0; k < NCH; ++k) a += hid[gg * NCH + k] * W2[k * 10 + o];
        outp[gg * 10 + o] = a;
    }
}

__global__ void zero_f32(float* __restrict__ p, long long n) {
    long long t = (long long)blockIdx.x * blockDim.x + threadIdx.x;
    if (t < n) p[t] = 0.f;
}

extern "C" void kernel_launch(void* const* d_in, const int* in_sizes, int n_in,
                              void* d_out, int out_size, void* d_ws, size_t ws_size,
                              hipStream_t stream) {
    (void)in_sizes; (void)n_in; (void)out_size; (void)ws_size;
    const float* x     = (const float*)d_in[0];
    const int*   ei    = (const int*)  d_in[1];
    const int*   batch = (const int*)  d_in[2];
    const float* W0    = (const float*)d_in[3];
    const float* as0   = (const float*)d_in[4];
    const float* ad0   = (const float*)d_in[5];
    const float* b0    = (const float*)d_in[6];
    const float* W1    = (const float*)d_in[7];
    const float* as1   = (const float*)d_in[8];
    const float* ad1   = (const float*)d_in[9];
    const float* b1    = (const float*)d_in[10];
    const float* W2    = (const float*)d_in[11];
    const float* as2   = (const float*)d_in[12];
    const float* ad2   = (const float*)d_in[13];
    const float* b2    = (const float*)d_in[14];
    const float* mW1   = (const float*)d_in[15];
    const float* mb1   = (const float*)d_in[16];
    const float* mW2   = (const float*)d_in[17];
    const float* mb2   = (const float*)d_in[18];
    float* out = (float*)d_out;

    // workspace layout (float units)
    float* ws = (float*)d_ws;
    const long long NF = (long long)NNODES * FDIM;
    float*    buf1  = ws;                        // h = X @ W  (f32)
    float*    buf2  = ws + NF;                   // aggregated output (zeroed/layer)
    unsigned* mmax  = (unsigned*)(ws + 2 * NF);  // N*4 (ordered-uint max)
    float*    den   = ws + 2 * NF + (long long)NNODES * 4;
    float*    a_src = ws + 2 * NF + (long long)NNODES * 8;
    float*    a_dst = ws + 2 * NF + (long long)NNODES * 12;
    float*    ex    = ws + 2 * NF + (long long)NNODES * 16;     // EPRIME*4
    float*    psum  = ex + (long long)EPRIME * 4;               // G*32
    float*    pcnt  = psum + NGRAPH * NCH;                      // G
    unsigned short* actbf = (unsigned short*)(pcnt + 16);       // N*128 bf16
    unsigned short* Bpack = actbf + NF;                         // 16384 bf16

    const int T = 256;
    const long long zlen = NF + (long long)NNODES * 8;   // buf2 + mmax + den
    const int zgrid = (int)((zlen + T - 1) / T);
    const int ngrid4   = (NNODES * 4 + T - 1) / T;
    const int egrid    = (EPRIME + T - 1) / T;
    const int agggrid  = (int)(((long long)EPRIME * 32 + T - 1) / T);
    const int ngrid128 = (int)((NF + T - 1) / T);
    const int ngrid32  = (NNODES * NCH + T - 1) / T;

    auto run_tail = [&](const float* as, const float* ad,
                        const float* bias, bool last) {
        gat_attn_coef<<<ngrid4, T, 0, stream>>>(buf1, as, ad, a_src, a_dst);
        edge_max_k<<<egrid, T, 0, stream>>>(ei, a_src, a_dst, mmax);
        edge_exp_k<<<egrid, T, 0, stream>>>(ei, a_src, a_dst, mmax, den, ex);
        edge_agg_k<<<agggrid, T, 0, stream>>>(ei, buf1, ex, den, buf2);
        if (!last) {
            act_elu_bias_bf16<<<ngrid128, T, 0, stream>>>(buf2, bias, actbf);
        } else {
            zero_f32<<<1, 128, 0, stream>>>(psum, NGRAPH * NCH + NGRAPH);
            final_pool<<<ngrid32, T, 0, stream>>>(buf2, bias, batch, psum, pcnt);
        }
    };

    // ---- layer 0 (K=3, f32 input) ----
    zero_f32<<<zgrid, T, 0, stream>>>(buf2, zlen);
    matmul_l0_wmma<<<NNODES / 16, T, 0, stream>>>(x, W0, buf1);
    run_tail(as0, ad0, b0, false);

    // ---- layer 1 (K=128, bf16 act, packed W staged via TDM) ----
    zero_f32<<<zgrid, T, 0, stream>>>(buf2, zlen);
    pack_W_bf16<<<4, 256, 0, stream>>>(W1, Bpack);
    matmul_l12_wmma<<<NNODES / (16 * 8), T, 0, stream>>>(actbf, Bpack, buf1);
    run_tail(as1, ad1, b1, false);

    // ---- layer 2 (K=128) ----
    zero_f32<<<zgrid, T, 0, stream>>>(buf2, zlen);
    pack_W_bf16<<<4, 256, 0, stream>>>(W2, Bpack);
    matmul_l12_wmma<<<NNODES / (16 * 8), T, 0, stream>>>(actbf, Bpack, buf1);
    run_tail(as2, ad2, b2, true);

    mlp_head<<<1, 64, 0, stream>>>(psum, pcnt, mW1, mb1, mW2, mb2, out);
}
